// MemoryEfficientAttention_90117003805176
// MI455X (gfx1250) — matrix operand order repaired and checked by reading, not compiled, not run on hardware
//
#include <hip/hip_runtime.h>
#include <hip/hip_bf16.h>

// ---------------------------------------------------------------------------
// MI455X (gfx1250) multi-head attention: bf16 WMMA + TDM (tensor_load_to_lds)
// B=2, L=2048, D=1024, H=16, HD=64.
// ---------------------------------------------------------------------------

typedef __attribute__((ext_vector_type(16))) __bf16 v16bf;
typedef __attribute__((ext_vector_type(8)))  __bf16 v8bf;
typedef __attribute__((ext_vector_type(8)))  float  v8f;
typedef __attribute__((ext_vector_type(4)))  float  v4f;
typedef unsigned int v4u __attribute__((ext_vector_type(4)));
typedef int          v8i_ __attribute__((ext_vector_type(8)));
typedef int          v4i_ __attribute__((ext_vector_type(4)));

#define B_      2
#define L_      2048
#define D_      1024
#define H_      16
#define HD_     64
#define NHEADS  (B_ * H_)    // 32
#define MROWS   (B_ * L_)    // 4096
// softmax computed in base-2 domain: exp(s/8) == exp2(s * 0.125 * log2 e)
#define SCALE_LOG2 0.1803368801111244f

#define LDSS 40              // padded LDS row stride (halves): 64B data + 16B pad

// -------------------------- small device helpers ---------------------------

__device__ __forceinline__ v8f zero8() {
    v8f z;
#pragma unroll
    for (int i = 0; i < 8; ++i) z[i] = 0.0f;
    return z;
}

__device__ __forceinline__ v16bf load16(const __bf16* p0, const __bf16* p1) {
    v8bf a = *(const v8bf*)p0;
    v8bf b = *(const v8bf*)p1;
    v16bf r;
#pragma unroll
    for (int i = 0; i < 8; ++i) { r[i] = a[i]; r[i + 8] = b[i]; }
    return r;
}

__device__ __forceinline__ v8f wmma_bf16(v16bf a, v16bf b, v8f c) {
    return __builtin_amdgcn_wmma_f32_16x16x32_bf16(
        false, a, false, b, (short)0, c, false, false);
}

// A-matrix 16x32 bf16 fragment (ISA 7.12.2): lane row M=lane&15,
// k = (i&7) + (i>=8?16:0) + 8*(lane>>4). Row-major source, stride in halves.
__device__ __forceinline__ v16bf load_a_frag(const __bf16* base, int row, int stride) {
    const int l = threadIdx.x & 31;
    const __bf16* p = base + (row + (l & 15)) * stride + (l >> 4) * 8;
    return load16(p, p + 16);
}

// B-matrix 32x16 bf16 fragment: lane column N=lane&15, k = (i&7) + (i>=8?16:0) + 8*(lane>>4).
// Source is the TRANSPOSED operand (rows indexed by N).
__device__ __forceinline__ v16bf load_b_frag(const __bf16* base, int nrow, int stride) {
    const int l = threadIdx.x & 31;
    const __bf16* p = base + (nrow + (l & 15)) * stride + (l >> 4) * 8;
    return load16(p, p + 16);
}

__device__ __forceinline__ float rowmax16(float v) {
#pragma unroll
    for (int m = 1; m < 16; m <<= 1) v = fmaxf(v, __shfl_xor(v, m, 32));
    return v;
}
__device__ __forceinline__ float rowsum16(float v) {
#pragma unroll
    for (int m = 1; m < 16; m <<= 1) v += __shfl_xor(v, m, 32);
    return v;
}

// --------------------------- TDM tile load (2D) -----------------------------
// Loads a [128 rows x 32 halves] bf16 tile (row stride 1024 halves in memory)
// into LDS with 16B padding after each 64B row -> LDS row stride = 40 halves.
// Descriptor packing per CDNA5 ISA 8.3/8.4 (D# groups 0/1; groups 2/3 unused).
__device__ __forceinline__ void tdm_load_tile(const __bf16* gsrc, unsigned lds_addr) {
    const unsigned long long ga = (unsigned long long)(size_t)gsrc;
    v4u g0;
    g0[0] = 1u;                                   // count=1, user descriptor
    g0[1] = lds_addr;                             // LDS byte address
    g0[2] = (unsigned)ga;                         // global_addr[31:0]
    g0[3] = ((unsigned)(ga >> 32) & 0x01FFFFFFu)  // global_addr[56:32]
          | 0x80000000u;                          // type=2 ("image")
    v8i_ g1;
    g1[0] = (int)((1u << 16)     // data_size = 2 bytes
                | (1u << 20)     // pad_enable
                | (3u << 22)     // pad_interval: 16 DWORDs (= 64B row)
                | (3u << 25));   // pad_amount:   4 DWORDs (= 16B pad)
    g1[1] = (int)(1024u << 16);              // tensor_dim0 = 1024 (lo16 @ [63:48])
    g1[2] = 0;                               // dim0 hi16=0 | tensor_dim1 lo16 = 0
    g1[3] = (int)(0x10u | (32u << 16));      // tensor_dim1 = 1<<20 (hi16) | tile_dim0=32
    g1[4] = 128;                             // tile_dim1 = 128, tile_dim2 = 0
    g1[5] = 1024;                            // tensor_dim0_stride lo32
    g1[6] = 0;                               // stride hi | dim1_stride lo (unused, 2D)
    g1[7] = 0;
    v4i_ z4; z4[0] = z4[1] = z4[2] = z4[3] = 0;
#if defined(__clang_major__) && (__clang_major__ >= 23)
    v8i_ z8;
#pragma unroll
    for (int i = 0; i < 8; ++i) z8[i] = 0;
    __builtin_amdgcn_tensor_load_to_lds(g0, g1, z4, z4, z8, 0);
#else
    __builtin_amdgcn_tensor_load_to_lds(g0, g1, z4, z4, 0);
#endif
}

// ----------------------- fp32 -> bf16 streaming convert ---------------------
__global__ __launch_bounds__(256) void cvt_bf16_kernel(
        const float* __restrict__ src, __bf16* __restrict__ dst) {
    const int i = (blockIdx.x * 256 + threadIdx.x) * 8;
    v4f f0 = *(const v4f*)(src + i);
    v4f f1 = *(const v4f*)(src + i + 4);
    v8bf h;
#pragma unroll
    for (int j = 0; j < 4; ++j) { h[j] = (__bf16)f0[j]; h[j + 4] = (__bf16)f1[j]; }
    *(v8bf*)(dst + i) = h;
}

// ------------------- weight transpose + fp32 -> bf16 prep -------------------
__global__ __launch_bounds__(256) void transpose_convert_kernel(
        const float* __restrict__ W, __bf16* __restrict__ Wt) {
    __shared__ float tile[32][33];
    const int bx = blockIdx.x * 32;   // n
    const int by = blockIdx.y * 32;   // k
    const int tx = threadIdx.x, ty = threadIdx.y;   // 32 x 8
#pragma unroll
    for (int j = 0; j < 32; j += 8)
        tile[ty + j][tx] = W[(by + ty + j) * D_ + bx + tx];
    __syncthreads();
#pragma unroll
    for (int j = 0; j < 32; j += 8)
        Wt[(bx + ty + j) * D_ + by + tx] = (__bf16)tile[tx][ty + j];
}

// -------------------- unified bf16 GEMM (TDM double-buffered) ---------------
// Y = A @ W + bias.   A: [4096][1024] bf16 row-major, Wt: [n][k] bf16.
// mode 0: dst_b[((b*16+h)*L + l)*64 + hd]   (Q / K head layout)
// mode 1: dst_b[((b*16+h)*64 + hd)*L + l]   (V transposed)
// mode 2: dst_f[m*1024 + n] fp32            (final projection)
// Block tile 128x128, 8 waves x (32x64), BK=32; tiles staged by the TDM.
__global__ __launch_bounds__(256) void gemm_bf16_kernel(
        const __bf16* __restrict__ A, const __bf16* __restrict__ Wt,
        const float* __restrict__ bias, __bf16* __restrict__ dst_b,
        float* __restrict__ dst_f, int mode) {
    __shared__ __bf16 smem[4 * 128 * LDSS];   // {A0, W0, A1, W1}
    const int TILE = 128 * LDSS;

    const int tid = threadIdx.x;
    const int l = tid & 31, w = tid >> 5;
    const int ln = l & 15, hl = l >> 4;
    const int wrow = w & 3, wcol = w >> 2;
    const int m0 = blockIdx.x * 128;
    const int n0 = blockIdx.y * 128;

    const __bf16* Abase = A  + (size_t)m0 * D_;
    const __bf16* Wbase = Wt + (size_t)n0 * D_;
    unsigned ldsA[2], ldsW[2];
    ldsA[0] = (unsigned)(size_t)(smem);
    ldsW[0] = (unsigned)(size_t)(smem + TILE);
    ldsA[1] = (unsigned)(size_t)(smem + 2 * TILE);
    ldsW[1] = (unsigned)(size_t)(smem + 3 * TILE);

    v8f acc[2][4];
#pragma unroll
    for (int i = 0; i < 2; ++i)
#pragma unroll
        for (int j = 0; j < 4; ++j) acc[i][j] = zero8();

    // Prologue: stage k=0 into buffer 0.
    if (w == 0) {
        tdm_load_tile(Abase, ldsA[0]);
        tdm_load_tile(Wbase, ldsW[0]);
    }

    const int NSTAGE = D_ / 32;
    for (int s = 0; s < NSTAGE; ++s) {
        if (w == 0) {
            if (s + 1 < NSTAGE) {   // prefetch next stage into the other buffer
                tdm_load_tile(Abase + (s + 1) * 32, ldsA[(s + 1) & 1]);
                tdm_load_tile(Wbase + (s + 1) * 32, ldsW[(s + 1) & 1]);
                __builtin_amdgcn_s_wait_tensorcnt(2);  // stage s complete (in-order)
            } else {
                __builtin_amdgcn_s_wait_tensorcnt(0);
            }
        }
        __syncthreads();

        const __bf16* bufA = smem + (s & 1) * 2 * TILE;
        const __bf16* bufW = bufA + TILE;
        v16bf a0 = load_a_frag(bufA, wrow * 32,      LDSS);
        v16bf a1 = load_a_frag(bufA, wrow * 32 + 16, LDSS);
#pragma unroll
        for (int t = 0; t < 4; ++t) {
            v16bf b = load_b_frag(bufW, wcol * 64 + t * 16, LDSS);
            acc[0][t] = wmma_bf16(a0, b, acc[0][t]);
            acc[1][t] = wmma_bf16(a1, b, acc[1][t]);
        }
        __syncthreads();   // everyone done with buf (s&1) before TDM rewrites it
    }

    // Epilogue.
#pragma unroll
    for (int t = 0; t < 4; ++t) {
        const int nn = n0 + wcol * 64 + t * 16 + ln;
        const float bv = bias[nn];
        const int h  = nn >> 6;
        const int hd = nn & 63;
#pragma unroll
        for (int rg = 0; rg < 2; ++rg) {
#pragma unroll
            for (int r = 0; r < 8; ++r) {
                const int m = m0 + wrow * 32 + rg * 16 + r + 8 * hl;
                const float v = acc[rg][t][r] + bv;
                if (mode == 2) {
                    dst_f[m * D_ + nn] = v;
                } else {
                    const int b  = m >> 11;
                    const int lq = m & (L_ - 1);
                    if (mode == 0)
                        dst_b[((b * H_ + h) * L_ + lq) * HD_ + hd] = (__bf16)v;
                    else
                        dst_b[((b * H_ + h) * HD_ + hd) * L_ + lq] = (__bf16)v;
                }
            }
        }
    }
}

// ---------------------- flash attention (one (b,h) head) --------------------
// Q,K: [bh][L][64] bf16 ; Vt: [bh][64][L] bf16 ; out attn: [B*L][D] bf16.
// 8 waves x 16 query rows = 128 rows/block; keys streamed 32 at a time.
__global__ __launch_bounds__(256) void attn_kernel(
        const __bf16* __restrict__ Qb, const __bf16* __restrict__ Kb,
        const __bf16* __restrict__ Vt, __bf16* __restrict__ attn) {
    __shared__ __bf16 plds[8 * 16 * LDSS];   // per-wave P staging (C->A relayout)

    const int tid = threadIdx.x;
    const int l = tid & 31, w = tid >> 5;
    const int ln = l & 15, hl = l >> 4;
    const int hb = blockIdx.y;               // b*16 + h
    const int b = hb >> 4, h = hb & 15;

    const __bf16* Qh = Qb + (size_t)hb * L_ * HD_;
    const __bf16* Kh = Kb + (size_t)hb * L_ * HD_;
    const __bf16* Vh = Vt + (size_t)hb * HD_ * L_;
    const int q0 = blockIdx.x * 128 + w * 16;

    const __bf16* qrow = Qh + (q0 + ln) * HD_;
    v16bf qa[2];
#pragma unroll
    for (int c = 0; c < 2; ++c) {
        const __bf16* p = qrow + 32 * c + 16 * hl;
        qa[c] = load16(p, p + 8);
    }

    v8f o[4];
#pragma unroll
    for (int t = 0; t < 4; ++t) o[t] = zero8();
    float mrun[8], lrun[8];
#pragma unroll
    for (int r = 0; r < 8; ++r) { mrun[r] = -1e30f; lrun[r] = 0.0f; }

    __bf16* pw = plds + w * 16 * LDSS;

    for (int kb = 0; kb < L_; kb += 32) {
        v8f s0 = zero8(), s1 = zero8();
#pragma unroll
        for (int c = 0; c < 2; ++c) {
            v16bf bk0 = *(const v16bf*)(Kh + (kb +      ln) * HD_ + 32 * c + 16 * hl);
            v16bf bk1 = *(const v16bf*)(Kh + (kb + 16 + ln) * HD_ + 32 * c + 16 * hl);
            s0 = wmma_bf16(qa[c], bk0, s0);
            s1 = wmma_bf16(qa[c], bk1, s1);
        }

        // Online softmax in base-2 domain (v_exp_f32 is natively 2^x).
#pragma unroll
        for (int r = 0; r < 8; ++r) {
            const float t0 = s0[r] * SCALE_LOG2;
            const float t1 = s1[r] * SCALE_LOG2;
            const float rm = rowmax16(fmaxf(t0, t1));
            const float mn = fmaxf(mrun[r], rm);
            const float al = __builtin_amdgcn_exp2f(mrun[r] - mn);
            mrun[r] = mn;
            const float p0 = __builtin_amdgcn_exp2f(t0 - mn);
            const float p1 = __builtin_amdgcn_exp2f(t1 - mn);
            lrun[r] = lrun[r] * al + rowsum16(p0 + p1);
            const int rr = r + 8 * hl;
            pw[rr * LDSS + ln]      = (__bf16)p0;
            pw[rr * LDSS + 16 + ln] = (__bf16)p1;
#pragma unroll
            for (int t = 0; t < 4; ++t) o[t][r] *= al;
        }

        // P back in A-fragment layout (per-wave LDS region; wave-order DS).
        const __bf16* pp = pw + ln * LDSS + 16 * hl;
        v16bf pa = load16(pp, pp + 8);

#pragma unroll
        for (int t = 0; t < 4; ++t) {
            v16bf bv = *(const v16bf*)(Vh + (t * 16 + ln) * L_ + kb + 16 * hl);
            o[t] = wmma_bf16(pa, bv, o[t]);
        }
    }

    float inv[8];
#pragma unroll
    for (int r = 0; r < 8; ++r) inv[r] = 1.0f / lrun[r];
#pragma unroll
    for (int t = 0; t < 4; ++t) {
#pragma unroll
        for (int r = 0; r < 8; ++r) {
            const int lq = q0 + r + 8 * hl;
            attn[(b * L_ + lq) * D_ + h * HD_ + t * 16 + ln] = (__bf16)(o[t][r] * inv[r]);
        }
    }
}

// ------------------------------- host launcher ------------------------------

extern "C" void kernel_launch(void* const* d_in, const int* in_sizes, int n_in,
                              void* d_out, int out_size, void* d_ws, size_t ws_size,
                              hipStream_t stream) {
    const float* query = (const float*)d_in[0];
    const float* key_  = (const float*)d_in[1];
    const float* value = (const float*)d_in[2];
    const float* Wq = (const float*)d_in[3];
    const float* bq = (const float*)d_in[4];
    const float* Wk = (const float*)d_in[5];
    const float* bk = (const float*)d_in[6];
    const float* Wv = (const float*)d_in[7];
    const float* bv = (const float*)d_in[8];
    const float* Wo = (const float*)d_in[9];
    const float* bo = (const float*)d_in[10];
    float* out = (float*)d_out;

    // Workspace (56 MB):
    //  [0,8)   Xq bf16   (later reused as attn output)
    //  [8,16)  Xk bf16
    //  [16,24) Xv bf16
    //  [24,32) Wt q/k/v/o bf16 (2MB each)
    //  [32,56) Qb, Kb, Vt bf16 (8MB each)
    char* ws = (char*)d_ws;
    const size_t XB = (size_t)MROWS * D_ * 2;            // 8 MB
    const size_t WB = (size_t)D_ * D_ * 2;               // 2 MB
    __bf16* Xq  = (__bf16*)(ws);
    __bf16* Xk  = (__bf16*)(ws + XB);
    __bf16* Xv  = (__bf16*)(ws + 2 * XB);
    __bf16* wtq = (__bf16*)(ws + 3 * XB);
    __bf16* wtk = (__bf16*)(ws + 3 * XB + WB);
    __bf16* wtv = (__bf16*)(ws + 3 * XB + 2 * WB);
    __bf16* wto = (__bf16*)(ws + 3 * XB + 3 * WB);
    __bf16* Qb  = (__bf16*)(ws + 3 * XB + 4 * WB);
    __bf16* Kb  = (__bf16*)(ws + 4 * XB + 4 * WB);
    __bf16* Vtb = (__bf16*)(ws + 5 * XB + 4 * WB);
    __bf16* attn = Xq;   // Xq consumed by the Q GEMM before attention writes it

    const int cvtBlocks = (MROWS * D_) / (256 * 8);      // 2048
    cvt_bf16_kernel<<<cvtBlocks, 256, 0, stream>>>(query, Xq);
    cvt_bf16_kernel<<<cvtBlocks, 256, 0, stream>>>(key_,  Xk);
    cvt_bf16_kernel<<<cvtBlocks, 256, 0, stream>>>(value, Xv);

    const dim3 tgrid(D_ / 32, D_ / 32), tblk(32, 8);
    transpose_convert_kernel<<<tgrid, tblk, 0, stream>>>(Wq, wtq);
    transpose_convert_kernel<<<tgrid, tblk, 0, stream>>>(Wk, wtk);
    transpose_convert_kernel<<<tgrid, tblk, 0, stream>>>(Wv, wtv);
    transpose_convert_kernel<<<tgrid, tblk, 0, stream>>>(Wo, wto);

    const dim3 ggrid(MROWS / 128, D_ / 128);
    gemm_bf16_kernel<<<ggrid, 256, 0, stream>>>(Xq, wtq, bq, Qb,  nullptr, 0);
    gemm_bf16_kernel<<<ggrid, 256, 0, stream>>>(Xk, wtk, bk, Kb,  nullptr, 0);
    gemm_bf16_kernel<<<ggrid, 256, 0, stream>>>(Xv, wtv, bv, Vtb, nullptr, 1);

    attn_kernel<<<dim3(L_ / 128, NHEADS), 256, 0, stream>>>(Qb, Kb, Vtb, attn);

    gemm_bf16_kernel<<<ggrid, 256, 0, stream>>>(attn, wto, bo, nullptr, out, 2);
}
